// priorCorrelation_4054449127561
// MI455X (gfx1250) — matrix operand order repaired and checked by reading, not compiled, not run on hardware
//
#include <hip/hip_runtime.h>
#include <math.h>

typedef __attribute__((ext_vector_type(16))) __bf16 v16bf;
typedef __attribute__((ext_vector_type(8)))  __bf16 v8bf;
typedef __attribute__((ext_vector_type(8)))  float  v8f;

#define B_  128
#define T_  512
#define N_  512
#define KC  32      // K-chunk per LDS stage (one WMMA K=32 step)
#define KCP 40      // padded LDS row stride (bf16 elems), multiple of 8

// ---------------- Phase 1: per-column mean and raw sum of squares ----------------
__global__ __launch_bounds__(256) void corr_stats_kernel(const float* __restrict__ x,
                                                         float* __restrict__ mu,
                                                         float* __restrict__ sq) {
  int g = blockIdx.x * 256 + threadIdx.x;   // 0 .. B_*N_-1, coalesced over n
  int b = g >> 9;
  int n = g & (N_ - 1);
  const float* xp = x + (size_t)b * T_ * N_ + n;
  float s = 0.f, s2 = 0.f;
#pragma unroll 8
  for (int t = 0; t < T_; ++t) {
    float v = xp[(size_t)t * N_];
    s  += v;
    s2 += v * v;
  }
  mu[g] = s * (1.0f / (float)T_);
  sq[g] = s2;
}

// split fp32 into bf16 hi (RNE) + bf16 lo of the remainder
__device__ __forceinline__ void split_bf(float v, unsigned short& h, unsigned short& l) {
  unsigned u  = __float_as_uint(v);
  unsigned hu = (u + 0x7FFFu + ((u >> 16) & 1u)) >> 16;
  float fh = __uint_as_float(hu << 16);
  float r  = v - fh;                       // exactly representable
  unsigned ru = __float_as_uint(r);
  unsigned lu = (ru + 0x7FFFu + ((ru >> 16) & 1u)) >> 16;
  h = (unsigned short)hu;
  l = (unsigned short)lu;
}

// assemble v16bf fragment from two 16-byte LDS reads
__device__ __forceinline__ v16bf frag_load(const __bf16* p, int second_off_elems) {
  v8bf a = *(const v8bf*)(p);
  v8bf b = *(const v8bf*)(p + second_off_elems);
  return __builtin_shufflevector(a, b, 0,1,2,3,4,5,6,7,8,9,10,11,12,13,14,15);
}

// ---------------- Phase 2: batched SYRK (bf16x3 split) + normalize ----------------
__global__ __launch_bounds__(256) void corr_gemm_kernel(const float* __restrict__ x,
                                                        const float* __restrict__ mu,
                                                        const float* __restrict__ sq,
                                                        float* __restrict__ out) {
  __shared__ __align__(16) __bf16 AsH[128 * KCP];
  __shared__ __align__(16) __bf16 AsL[128 * KCP];
  __shared__ __align__(16) __bf16 BsH[128 * KCP];
  __shared__ __align__(16) __bf16 BsL[128 * KCP];
  __shared__ float dA[128];
  __shared__ float dB[128];

  const int bi = blockIdx.x;
  const int b  = bi >> 4;
  const int tm = (bi >> 2) & 3;
  const int tn = bi & 3;
  const int m0 = tm * 128, n0 = tn * 128;
  const float* xb = x + (size_t)b * T_ * N_;

  const int tid  = threadIdx.x;
  const int lane = tid & 31;
  const int wave = tid >> 5;      // 8 waves
  const int wr   = wave >> 1;     // 0..3 -> 32-row strip
  const int wc   = wave & 1;      // 0..1 -> 64-col strip
  const int half = (lane >> 4) & 1;
  const int l15  = lane & 15;

  // loader mapping: 128 columns x 2 row-teams, each thread handles 2 t-rows/step
  const int lcol = tid & 127;
  const int lrp  = (tid >> 7) * 2;
  const float muA = mu[b * N_ + m0 + lcol];
  const float muB = mu[b * N_ + n0 + lcol];
  const float* gA = xb + m0 + lcol;
  const float* gB = xb + n0 + lcol;

  // per-column scale inputs: d = sqrt(sum(x^2) - T*mu^2)
  if (tid < 128) {
    float m_ = mu[b * N_ + m0 + tid];
    dA[tid]  = sqrtf(sq[b * N_ + m0 + tid] - (float)T_ * m_ * m_);
  } else {
    int   c  = tid - 128;
    float m_ = mu[b * N_ + n0 + c];
    dB[c]    = sqrtf(sq[b * N_ + n0 + c] - (float)T_ * m_ * m_);
  }

  v8f acc[8];
  v8f zero = {};
#pragma unroll
  for (int i = 0; i < 8; ++i) acc[i] = zero;

  // register-prefetch pipeline buffers (chunk data lives in VGPRs one stage ahead)
  float ra[16], rb[16];

  // prologue: prefetch chunk 0
#pragma unroll
  for (int it = 0; it < 8; ++it) {
    int row = it * 4 + lrp;
    ra[2*it]   = gA[(size_t)(row    ) * N_];
    ra[2*it+1] = gA[(size_t)(row + 1) * N_];
    rb[2*it]   = gB[(size_t)(row    ) * N_];
    rb[2*it+1] = gB[(size_t)(row + 1) * N_];
  }

  for (int t0 = 0; t0 < T_; t0 += KC) {
    __syncthreads();   // previous chunk's LDS reads complete
    // ---- convert prefetched regs: center + bf16 hi/lo split -> LDS ----
#pragma unroll
    for (int it = 0; it < 8; ++it) {
      int row = it * 4 + lrp;
      unsigned short h0, l0, h1, l1;
      split_bf(ra[2*it]   - muA, h0, l0);
      split_bf(ra[2*it+1] - muA, h1, l1);
      *(unsigned int*)&AsH[lcol * KCP + row] = (unsigned)h0 | ((unsigned)h1 << 16);
      *(unsigned int*)&AsL[lcol * KCP + row] = (unsigned)l0 | ((unsigned)l1 << 16);
      split_bf(rb[2*it]   - muB, h0, l0);
      split_bf(rb[2*it+1] - muB, h1, l1);
      *(unsigned int*)&BsH[lcol * KCP + row] = (unsigned)h0 | ((unsigned)h1 << 16);
      *(unsigned int*)&BsL[lcol * KCP + row] = (unsigned)l0 | ((unsigned)l1 << 16);
    }
    __syncthreads();

    // ---- issue next chunk's global loads; they retire under the WMMAs below ----
    if (t0 + KC < T_) {
      int tn0 = t0 + KC;
#pragma unroll
      for (int it = 0; it < 8; ++it) {
        int row = tn0 + it * 4 + lrp;
        ra[2*it]   = gA[(size_t)(row    ) * N_];
        ra[2*it+1] = gA[(size_t)(row + 1) * N_];
        rb[2*it]   = gB[(size_t)(row    ) * N_];
        rb[2*it+1] = gB[(size_t)(row + 1) * N_];
      }
    }

    // ---- build fragments per ISA 16-bit layouts, issue WMMAs ----
    v16bf aH[2], aL[2], bH[4], bL[4];
#pragma unroll
    for (int ta = 0; ta < 2; ++ta) {
      int r = wr * 32 + ta * 16 + l15;                   // A: lanes 0-15 K{0-7,16-23}, 16-31 K{8-15,24-31}
      aH[ta] = frag_load(&AsH[r * KCP + half * 8], 16);
      aL[ta] = frag_load(&AsL[r * KCP + half * 8], 16);
    }
#pragma unroll
    for (int tb = 0; tb < 4; ++tb) {
      int c = wc * 64 + tb * 16 + l15;                   // B: lanes 0-15 K{0-7,16-23}, 16-31 K{8-15,24-31}
      bH[tb] = frag_load(&BsH[c * KCP + half * 8], 16);
      bL[tb] = frag_load(&BsL[c * KCP + half * 8], 16);
    }
#pragma unroll
    for (int ta = 0; ta < 2; ++ta)
#pragma unroll
      for (int tb = 0; tb < 4; ++tb) {
        int i = ta * 4 + tb;
        acc[i] = __builtin_amdgcn_wmma_f32_16x16x32_bf16(false, aH[ta], false, bH[tb],
                                                         (short)0, acc[i], false, false);
        acc[i] = __builtin_amdgcn_wmma_f32_16x16x32_bf16(false, aH[ta], false, bL[tb],
                                                         (short)0, acc[i], false, false);
        acc[i] = __builtin_amdgcn_wmma_f32_16x16x32_bf16(false, aL[ta], false, bH[tb],
                                                         (short)0, acc[i], false, false);
      }
  }
  __syncthreads();

  // ---- epilogue: correlation normalize, clip, abs, diag=1, nan->0 ----
  float* ob = out + (size_t)b * N_ * N_;
#pragma unroll
  for (int ta = 0; ta < 2; ++ta)
#pragma unroll
    for (int tb = 0; tb < 4; ++tb) {
      int i  = ta * 4 + tb;
      int nl = wc * 64 + tb * 16 + l15;
      float dn = dB[nl];
#pragma unroll
      for (int e = 0; e < 8; ++e) {
        int ml = wr * 32 + ta * 16 + half * 8 + e;       // C/D: VGPR e = M e (lanes 0-15) / 8+e (16-31)
        float dm = dA[ml];
        int gm = m0 + ml, gn = n0 + nl;
        float denom = dm * dn;
        float v = fabsf(acc[i][e]) / denom;
        float o = (gm == gn) ? 1.0f : ((denom > 0.0f) ? fminf(v, 1.0f) : 0.0f);
        ob[(size_t)gm * N_ + gn] = o;
      }
    }
}

extern "C" void kernel_launch(void* const* d_in, const int* in_sizes, int n_in,
                              void* d_out, int out_size, void* d_ws, size_t ws_size,
                              hipStream_t stream) {
  const float* x = (const float*)d_in[0];
  float* out = (float*)d_out;
  float* mu  = (float*)d_ws;             // B_*N_ floats
  float* sq  = mu + B_ * N_;             // B_*N_ floats (512 KB total workspace)

  corr_stats_kernel<<<(B_ * N_) / 256, 256, 0, stream>>>(x, mu, sq);
  corr_gemm_kernel<<<B_ * 16, 256, 0, stream>>>(x, mu, sq, out);
}
